// Casualselfatt_79688823210055
// MI455X (gfx1250) — matrix operand order repaired and checked by reading, not compiled, not run on hardware
//
#include <hip/hip_runtime.h>

// ---------------------------------------------------------------------------
// Types for CDNA5 WMMA (wave32): v_wmma_f32_16x16x32_bf16
// ---------------------------------------------------------------------------
typedef __bf16 bf16;
typedef __attribute__((ext_vector_type(16))) bf16  v16bf;
typedef __attribute__((ext_vector_type(8)))  bf16  v8bf;
typedef __attribute__((ext_vector_type(8)))  float v8f;

// fp32 -> bf16 (round to nearest even), as raw bits
__device__ __forceinline__ unsigned short f2bf(float f) {
    unsigned int u = __builtin_bit_cast(unsigned int, f);
    u += 0x7fffu + ((u >> 16) & 1u);
    return (unsigned short)(u >> 16);
}

// Build a v16bf fragment from two 16-byte groups (p0: elems 0..7, p1: 8..15)
__device__ __forceinline__ v16bf make_frag2(const unsigned short* p0,
                                            const unsigned short* p1) {
    v8bf a = *(const v8bf*)p0;
    v8bf b = *(const v8bf*)p1;
    v16bf r;
#pragma unroll
    for (int i = 0; i < 8; ++i) { r[i] = a[i]; r[8 + i] = b[i]; }
    return r;
}
// B-fragment for one lane: bf16 groups p[0..7] and p[16..23]
__device__ __forceinline__ v16bf make_frag16(const unsigned short* p) {
    return make_frag2(p, p + 16);
}

__device__ __forceinline__ v8f wmma_bf16(v16bf a, v16bf b, v8f c) {
    return __builtin_amdgcn_wmma_f32_16x16x32_bf16(
        /*neg_a=*/false, a, /*neg_b=*/false, b,
        /*c_mod=*/(short)0, c, /*reuse_a=*/false, /*reuse_b=*/false);
}

// ---------------------------------------------------------------------------
// Problem constants
// ---------------------------------------------------------------------------
#define BB   2
#define SS   2048
#define EE   1024
#define HH   16
#define DD   64
#define MM   (BB * SS)      // 4096
#define N3E  (3 * EE)       // 3072

// ---------------------------------------------------------------------------
// Converters
// ---------------------------------------------------------------------------
__global__ void cvt_kernel(const float* __restrict__ in,
                           unsigned short* __restrict__ out, int n) {
    int i = blockIdx.x * 256 + threadIdx.x;
    if (i < n) out[i] = f2bf(in[i]);
}

// in[K][N] (row-major) -> out[N][K] bf16 (n-major: contiguous K per row)
__global__ void cvt_t_kernel(const float* __restrict__ in,
                             unsigned short* __restrict__ out, int K, int N) {
    int i = blockIdx.x * 256 + threadIdx.x;
    if (i < K * N) {
        int nn = i / K, kk = i % K;
        out[i] = f2bf(in[kk * N + nn]);
    }
}

// ---------------------------------------------------------------------------
// Shared GEMM core: ping-pong double-buffered bf16 WMMA accumulation.
// Wave computes a 16x64 tile; K in steps of 32, unrolled 2x so the buffer
// swap is register renaming (no v_mov chains).
// ---------------------------------------------------------------------------
__device__ __forceinline__ void gemm_core(const unsigned short* __restrict__ arow,
                                          const unsigned short* const brow[4],
                                          int cbA, v8f acc[4]) {
    // set A <- k=0
    v16bf aA = make_frag2(arow + cbA, arow + cbA + 16);
    v16bf bA[4];
#pragma unroll
    for (int c = 0; c < 4; ++c) bA[c] = make_frag16(brow[c]);
    v16bf aB, bB[4];

    for (int kk = 0; kk <= EE - 128; kk += 64) {
        // prefetch next iteration's cachelines (global_prefetch_b8)
        __builtin_prefetch(arow + kk + 128, 0, 1);
#pragma unroll
        for (int c = 0; c < 4; ++c) __builtin_prefetch(brow[c] + kk + 128, 0, 1);

        // set B <- kk+32 (issue before consuming set A)
        aB = make_frag2(arow + kk + 32 + cbA, arow + kk + 32 + cbA + 16);
#pragma unroll
        for (int c = 0; c < 4; ++c) bB[c] = make_frag16(brow[c] + kk + 32);
#pragma unroll
        for (int c = 0; c < 4; ++c) acc[c] = wmma_bf16(aA, bA[c], acc[c]);

        // set A <- kk+64 (issue before consuming set B)
        aA = make_frag2(arow + kk + 64 + cbA, arow + kk + 64 + cbA + 16);
#pragma unroll
        for (int c = 0; c < 4; ++c) bA[c] = make_frag16(brow[c] + kk + 64);
#pragma unroll
        for (int c = 0; c < 4; ++c) acc[c] = wmma_bf16(aB, bB[c], acc[c]);
    }
    // tail: kk = EE-64 .. EE
    aB = make_frag2(arow + EE - 32 + cbA, arow + EE - 32 + cbA + 16);
#pragma unroll
    for (int c = 0; c < 4; ++c) bB[c] = make_frag16(brow[c] + EE - 32);
#pragma unroll
    for (int c = 0; c < 4; ++c) acc[c] = wmma_bf16(aA, bA[c], acc[c]);
#pragma unroll
    for (int c = 0; c < 4; ++c) acc[c] = wmma_bf16(aB, bB[c], acc[c]);
}

// ---------------------------------------------------------------------------
// Kernel 1: QKV GEMM  C[4096,3072] = xb[4096,1024] @ Wqkv + b
// Epilogue scatters into per-head Qp[bh][s][d], Kp[bh][s][d], Vt[bh][d][s]
// using the reference's reshape mapping: h=s2/128, s=(s2%128)*16+e2/64, d=e2%64
// ---------------------------------------------------------------------------
__global__ void __launch_bounds__(128)
qkv_gemm_kernel(const unsigned short* __restrict__ xb,
                const unsigned short* __restrict__ wt,   // [3072][1024] n-major
                const float* __restrict__ bias,
                unsigned short* __restrict__ Qp,
                unsigned short* __restrict__ Kp,
                unsigned short* __restrict__ Vt) {
    const int lane = threadIdx.x & 31;
    const int wave = threadIdx.x >> 5;
    const int m0 = blockIdx.x * 64 + wave * 16;
    const int n0 = blockIdx.y * 64;
    const int rowA = m0 + (lane & 15);
    const int cbA  = (lane < 16) ? 0 : 8;
    const int nB   = lane & 15;
    const int kbB  = (lane < 16) ? 0 : 8;

    v8f acc[4];
#pragma unroll
    for (int c = 0; c < 4; ++c)
#pragma unroll
        for (int r = 0; r < 8; ++r) acc[c][r] = 0.0f;

    const unsigned short* arow = xb + (size_t)rowA * EE;
    const unsigned short* brow[4];
#pragma unroll
    for (int c = 0; c < 4; ++c)
        brow[c] = wt + (size_t)(n0 + c * 16 + nB) * EE + kbB;

    gemm_core(arow, brow, cbA, acc);

    // epilogue: bias + bf16 + head-relabel scatter
#pragma unroll
    for (int c = 0; c < 4; ++c) {
        int n = n0 + c * 16 + nB;
        int part = n >> 10;       // 0=Q 1=K 2=V
        int e2 = n & 1023;
        float bv = bias[n];
#pragma unroll
        for (int r = 0; r < 8; ++r) {
            int m  = m0 + ((lane < 16) ? r : 8 + r);
            int b_ = m >> 11;
            int s2 = m & 2047;
            int h  = s2 >> 7;
            int s  = (s2 & 127) * 16 + (e2 >> 6);
            int d  = e2 & 63;
            int bh = b_ * HH + h;
            unsigned short hv = f2bf(acc[c][r] + bv);
            if (part == 0)      Qp[((size_t)bh * SS + s) * DD + d] = hv;
            else if (part == 1) Kp[((size_t)bh * SS + s) * DD + d] = hv;
            else                Vt[((size_t)bh * DD + d) * SS + s] = hv;
        }
    }
}

// ---------------------------------------------------------------------------
// Kernel 2: causal flash attention, one wave per 16-row query tile.
// 4 waves / block, NO block barriers (causal trip counts differ per wave).
// V fragments hoisted ahead of QK/softmax; next tile prefetched into WGP$.
// ---------------------------------------------------------------------------
__global__ void __launch_bounds__(128)
attn_kernel(const unsigned short* __restrict__ Qp,
            const unsigned short* __restrict__ Kp,
            const unsigned short* __restrict__ Vt,
            unsigned short* __restrict__ AO) {
    __shared__ __align__(16) unsigned short pst[4][16][32];

    const int lane  = threadIdx.x & 31;
    const int wave  = threadIdx.x >> 5;
    const int bh    = blockIdx.y;                 // 0..31
    const int qbase = blockIdx.x * 64 + wave * 16;

    const unsigned short* Q = Qp + (size_t)bh * SS * DD;
    const unsigned short* K = Kp + (size_t)bh * SS * DD;
    const unsigned short* V = Vt + (size_t)bh * DD * SS;

    const int cbA = (lane < 16) ? 0 : 8;
    const int nB  = lane & 15;
    const int kbB = (lane < 16) ? 0 : 8;
    const int rowq = qbase + nB;

    // Q fragments: 16x64 split into two 16x32 A-fragments (held in VGPRs)
    v16bf q0 = make_frag2(Q + rowq * DD + 0  + cbA, Q + rowq * DD + 0  + cbA + 16);
    v16bf q1 = make_frag2(Q + rowq * DD + 32 + cbA, Q + rowq * DD + 32 + cbA + 16);

    float mrow[8], lrow[8];
    v8f o[4];
#pragma unroll
    for (int r = 0; r < 8; ++r) { mrow[r] = -3.0e38f; lrow[r] = 0.0f; }
#pragma unroll
    for (int c = 0; c < 4; ++c)
#pragma unroll
        for (int r = 0; r < 8; ++r) o[c][r] = 0.0f;

    const int nkt = (qbase + 16 + 31) >> 5;       // causal: keys <= qbase+15
    for (int t = 0; t < nkt; ++t) {
        const int kt = t * 32;

        // prefetch next key-step K/V rows into WGP$ (global_prefetch_b8)
        if (t + 1 < nkt) {
            const int kn = kt + 32;
            __builtin_prefetch(K + (size_t)(kn + nB) * DD, 0, 1);
            __builtin_prefetch(K + (size_t)(kn + 16 + nB) * DD, 0, 1);
            __builtin_prefetch(V + (size_t)nB * SS + kn, 0, 1);
            __builtin_prefetch(V + (size_t)(32 + nB) * SS + kn, 0, 1);
        }

        // V fragments for this step: independent of S, issue early to overlap
        v16bf vb[4];
#pragma unroll
        for (int c = 0; c < 4; ++c)
            vb[c] = make_frag16(V + (size_t)(c * 16 + nB) * SS + kt + kbB);

        // S = Q * K^T   (two 16x16 tiles over 32 keys, K-dim D=64 in 2 chunks)
        v8f s0, s1;
#pragma unroll
        for (int r = 0; r < 8; ++r) { s0[r] = 0.0f; s1[r] = 0.0f; }
        {
            const unsigned short* kp0 = K + (size_t)(kt + nB) * DD;
            const unsigned short* kp1 = K + (size_t)(kt + 16 + nB) * DD;
            v16bf kb00 = make_frag16(kp0 + 0  + kbB);
            v16bf kb01 = make_frag16(kp0 + 32 + kbB);
            v16bf kb10 = make_frag16(kp1 + 0  + kbB);
            v16bf kb11 = make_frag16(kp1 + 32 + kbB);
            s0 = wmma_bf16(q0, kb00, s0);
            s0 = wmma_bf16(q1, kb01, s0);
            s1 = wmma_bf16(q0, kb10, s1);
            s1 = wmma_bf16(q1, kb11, s1);
        }

        // online softmax (C/D layout: lane%16 = column, reg/half = row)
        const float scale = 0.125f;               // 1/sqrt(64)
        const int col0 = kt + nB;
        const int col1 = kt + 16 + nB;
#pragma unroll
        for (int r = 0; r < 8; ++r) {
            int qi = qbase + ((lane < 16) ? r : 8 + r);
            float v0 = s0[r] * scale; if (col0 > qi) v0 = -1.0e30f;
            float v1 = s1[r] * scale; if (col1 > qi) v1 = -1.0e30f;
            float tmax = fmaxf(v0, v1);
            tmax = fmaxf(tmax, __shfl_xor(tmax, 1, 32));
            tmax = fmaxf(tmax, __shfl_xor(tmax, 2, 32));
            tmax = fmaxf(tmax, __shfl_xor(tmax, 4, 32));
            tmax = fmaxf(tmax, __shfl_xor(tmax, 8, 32));
            float mnew = fmaxf(mrow[r], tmax);
            float p0 = __expf(v0 - mnew);
            float p1 = __expf(v1 - mnew);
            float rs = p0 + p1;
            rs += __shfl_xor(rs, 1, 32);
            rs += __shfl_xor(rs, 2, 32);
            rs += __shfl_xor(rs, 4, 32);
            rs += __shfl_xor(rs, 8, 32);
            float alpha = __expf(mrow[r] - mnew);
            mrow[r] = mnew;
            lrow[r] = lrow[r] * alpha + rs;
#pragma unroll
            for (int c = 0; c < 4; ++c) o[c][r] *= alpha;
            // stage P (D-layout -> LDS) for transpose into A-layout
            int prow = (lane < 16) ? r : 8 + r;
            pst[wave][prow][nB]      = f2bf(p0);
            pst[wave][prow][16 + nB] = f2bf(p1);
        }
        asm volatile("s_wait_dscnt 0" ::: "memory");  // per-wave LDS RAW fence

        // P as A-fragment (16 rows x 32 keys)
        const int prow2 = lane & 15;
        v16bf pf = make_frag2(&pst[wave][prow2][cbA], &pst[wave][prow2][cbA + 16]);
        asm volatile("" ::: "memory");                 // keep loads before next-iter stores

        // O += P * V  (V fragments already in registers)
#pragma unroll
        for (int c = 0; c < 4; ++c) o[c] = wmma_bf16(pf, vb[c], o[c]);
    }

    // write attention output into proj-GEMM row layout: AO[b*S + s][h*64 + d]
    const int b_ = bh >> 4, h = bh & 15;
#pragma unroll
    for (int r = 0; r < 8; ++r) {
        float inv = 1.0f / lrow[r];
        int qi = qbase + ((lane < 16) ? r : 8 + r);
#pragma unroll
        for (int c = 0; c < 4; ++c) {
            int d = c * 16 + nB;
            AO[((size_t)b_ * SS + qi) * EE + h * DD + d] = f2bf(o[c][r] * inv);
        }
    }
}

// ---------------------------------------------------------------------------
// Kernel 3: projection GEMM  out[4096,1024] = AO @ Wproj + b  (fp32 out)
// ---------------------------------------------------------------------------
__global__ void __launch_bounds__(128)
proj_gemm_kernel(const unsigned short* __restrict__ ab,
                 const unsigned short* __restrict__ wt,  // [1024][1024] n-major
                 const float* __restrict__ bias,
                 float* __restrict__ out) {
    const int lane = threadIdx.x & 31;
    const int wave = threadIdx.x >> 5;
    const int m0 = blockIdx.x * 64 + wave * 16;
    const int n0 = blockIdx.y * 64;
    const int rowA = m0 + (lane & 15);
    const int cbA  = (lane < 16) ? 0 : 8;
    const int nB   = lane & 15;
    const int kbB  = (lane < 16) ? 0 : 8;

    v8f acc[4];
#pragma unroll
    for (int c = 0; c < 4; ++c)
#pragma unroll
        for (int r = 0; r < 8; ++r) acc[c][r] = 0.0f;

    const unsigned short* arow = ab + (size_t)rowA * EE;
    const unsigned short* brow[4];
#pragma unroll
    for (int c = 0; c < 4; ++c)
        brow[c] = wt + (size_t)(n0 + c * 16 + nB) * EE + kbB;

    gemm_core(arow, brow, cbA, acc);

#pragma unroll
    for (int c = 0; c < 4; ++c) {
        int n = n0 + c * 16 + nB;
        float bv = bias[n];
#pragma unroll
        for (int r = 0; r < 8; ++r) {
            int m = m0 + ((lane < 16) ? r : 8 + r);
            out[(size_t)m * EE + n] = acc[c][r] + bv;
        }
    }
}

// ---------------------------------------------------------------------------
// Host launcher
// ---------------------------------------------------------------------------
extern "C" void kernel_launch(void* const* d_in, const int* in_sizes, int n_in,
                              void* d_out, int out_size, void* d_ws, size_t ws_size,
                              hipStream_t stream) {
    const float* x     = (const float*)d_in[0];
    const float* Wqkv  = (const float*)d_in[1];
    const float* bqkv  = (const float*)d_in[2];
    const float* Wproj = (const float*)d_in[3];
    const float* bproj = (const float*)d_in[4];
    float* out = (float*)d_out;

    char* ws = (char*)d_ws;
    size_t off = 0;
    auto alloc = [&](size_t bytes) {
        void* p = ws + off;
        off += (bytes + 255) & ~(size_t)255;
        return p;
    };
    unsigned short* xb  = (unsigned short*)alloc((size_t)MM * EE * 2);        // 8 MB
    unsigned short* wqt = (unsigned short*)alloc((size_t)N3E * EE * 2);       // 6 MB
    unsigned short* wpt = (unsigned short*)alloc((size_t)EE * EE * 2);        // 2 MB
    unsigned short* Qp  = (unsigned short*)alloc((size_t)BB * HH * SS * DD * 2); // 8 MB
    unsigned short* Kp  = (unsigned short*)alloc((size_t)BB * HH * SS * DD * 2); // 8 MB
    unsigned short* Vt  = (unsigned short*)alloc((size_t)BB * HH * DD * SS * 2); // 8 MB
    unsigned short* AO  = (unsigned short*)alloc((size_t)MM * EE * 2);        // 8 MB

    int nx = MM * EE;
    cvt_kernel<<<(nx + 255) / 256, 256, 0, stream>>>(x, xb, nx);
    int nwq = EE * N3E;
    cvt_t_kernel<<<(nwq + 255) / 256, 256, 0, stream>>>(Wqkv, wqt, EE, N3E);
    int nwp = EE * EE;
    cvt_t_kernel<<<(nwp + 255) / 256, 256, 0, stream>>>(Wproj, wpt, EE, EE);

    qkv_gemm_kernel<<<dim3(MM / 64, N3E / 64), 128, 0, stream>>>(xb, wqt, bqkv, Qp, Kp, Vt);
    attn_kernel<<<dim3(SS / 64, BB * HH), 128, 0, stream>>>(Qp, Kp, Vt, AO);
    proj_gemm_kernel<<<dim3(MM / 64, EE / 64), 128, 0, stream>>>(AO, wpt, bproj, out);
}
